// Decoder_70549132804202
// MI455X (gfx1250) — compile-verified, matrix operand order checked by reading, not run on hardware
//
#include <hip/hip_runtime.h>
#include <hip/hip_bf16.h>
#include <math.h>

#define V_ 32000
#define E_ 512
#define H_ 512
#define L_ 4
#define B_ 32
#define T_ 64

typedef unsigned short u16;
typedef __attribute__((ext_vector_type(16))) __bf16 v16bf;
typedef __attribute__((ext_vector_type(8)))  float  v8f;
typedef __attribute__((ext_vector_type(4)))  unsigned u32x4;
typedef __attribute__((ext_vector_type(4)))  int      i32x4;

struct F32B { u32x4 lo, hi; };   // 32-byte fragment container (two 16B chunks)

// ---------------- CDNA5 async global->LDS helpers (ASYNCcnt path) ----------
#if __has_builtin(__builtin_amdgcn_global_load_async_to_lds_b128)
#define HAS_ASYNC_LDS 1
typedef __attribute__((address_space(1))) i32x4* as1_v4i;
typedef __attribute__((address_space(3))) i32x4* as3_v4i;
#else
#define HAS_ASYNC_LDS 0
#endif

__device__ __forceinline__ void async_copy16(void* lds_dst, const void* gsrc) {
#if HAS_ASYNC_LDS
    __builtin_amdgcn_global_load_async_to_lds_b128(
        (as1_v4i)(gsrc), (as3_v4i)(lds_dst), 0, 0);
#else
    *(u32x4*)lds_dst = *(const u32x4*)gsrc;
#endif
}

__device__ __forceinline__ void async_wait0() {
#if HAS_ASYNC_LDS
#if __has_builtin(__builtin_amdgcn_s_wait_asynccnt)
    __builtin_amdgcn_s_wait_asynccnt(0);
#else
    asm volatile("s_wait_asynccnt 0" ::: "memory");
#endif
#endif
}

__device__ __forceinline__ u16 f2bf(float f) {
    union { float f; unsigned u; } x; x.f = f;
    unsigned r = x.u + 0x7FFFu + ((x.u >> 16) & 1u);  // round-to-nearest-even
    return (u16)(r >> 16);
}

__device__ __forceinline__ v16bf load_frag_lds(const u16* base, int kof) {
    // elements [kof, kof+8) and [kof+16, kof+24) of a 32-element K row
    F32B f;
    f.lo = *(const u32x4*)(base + kof);
    f.hi = *(const u32x4*)(base + kof + 16);
    return __builtin_bit_cast(v16bf, f);
}

__device__ __forceinline__ float sigmoidf_(float x) {
    return 1.0f / (1.0f + __expf(-x));
}

// ---------------------------------------------------------------- embedding
__global__ void embed_kernel(const float* __restrict__ emb,
                             const int* __restrict__ idx,
                             float* __restrict__ x) {
    int row = blockIdx.x;                 // b*T + t  (0..2047)
    int tok = idx[row];
    const float4* src = (const float4*)(emb + (size_t)tok * E_);
    float4* dst = (float4*)(x + (size_t)row * E_);
    dst[threadIdx.x] = src[threadIdx.x];  // 128 threads * float4 = 512 floats
}

// ------------------------------------------------- weight prep (f32 -> bf16)
__global__ void prep_wlstm(const float* __restrict__ W_ih,
                           const float* __restrict__ W_hh,
                           u16* __restrict__ Wl) {
    size_t e = ((size_t)blockIdx.x * blockDim.x + threadIdx.x) * 4;
    int k = (int)(e & 1023);
    int n = (int)((e >> 10) & 2047);
    int l = (int)(e >> 21);
    const float* src = (k < 512)
        ? (W_ih + (((size_t)l * 2048 + n) * 512 + k))
        : (W_hh + (((size_t)l * 2048 + n) * 512 + (k - 512)));
    u16* dst = Wl + e;
    #pragma unroll
    for (int j = 0; j < 4; ++j) dst[j] = f2bf(src[j]);
}

__global__ void prep_wcls(const float* __restrict__ W, u16* __restrict__ Wb) {
    size_t e = ((size_t)blockIdx.x * blockDim.x + threadIdx.x) * 4;
    #pragma unroll
    for (int j = 0; j < 4; ++j) Wb[e + j] = f2bf(W[e + j]);
}

// ------------------------------------------------------------- LSTM layer
__global__ void lstm_layer_kernel(const u16* __restrict__ Wl,      // [L][2048][1024] bf16
                                  const float* __restrict__ b_ih,  // [L][2048]
                                  const float* __restrict__ b_hh,  // [L][2048]
                                  const float* __restrict__ h_in,  // [L][B][H]
                                  float* __restrict__ h_out,       // [L][B][H]
                                  float* __restrict__ c,           // [L][B][H]
                                  const float* __restrict__ inp, int inp_stride,
                                  float* __restrict__ outp, int out_stride,
                                  int layer) {
    __shared__ u16   As[32 * 32];        // [row][k]            2 KB
    __shared__ u16   Ws[4 * 32 * 32];    // [strip*32+col][k]   8 KB
    __shared__ float Gs[32 * 128];       // gates [row][strip*32+col] 16 KB

    const int tid  = threadIdx.x;
    const int lane = tid & 31;
    const int wave = tid >> 5;
    const int bk   = blockIdx.x;         // 16 blocks
    const int j0   = bk * 32;

    const float* hL  = h_in  + ((size_t)layer * B_) * H_;
    float*       hLw = h_out + ((size_t)layer * B_) * H_;
    float*       cL  = c     + ((size_t)layer * B_) * H_;

    const int mTile = wave >> 2;         // 0..1
    const int nPair = wave & 3;          // 0..3 -> nTiles {2p, 2p+1}

    v8f acc[2];
    #pragma unroll
    for (int q = 0; q < 2; ++q) {
        int nT = nPair * 2 + q;
        int strip = nT >> 1;
        int n = strip * 512 + j0 + (nT & 1) * 16 + (lane & 15);
        float bias = b_ih[layer * 2048 + n] + b_hh[layer * 2048 + n];
        #pragma unroll
        for (int i = 0; i < 8; ++i) acc[q][i] = bias;
    }

    const u16* Wbase = Wl + ((size_t)layer * 2048) * 1024;

    for (int kb = 0; kb < 32; ++kb) {
        const int k0 = kb * 32;
        // stage W panel via async global->LDS (pure bf16 copy, no VGPR round-trip)
        {
            int r     = tid >> 1;              // 0..127
            int strip = r >> 5;
            int col   = r & 31;
            int koff  = (tid & 1) * 16;
            int n     = strip * 512 + j0 + col;
            const u16* src = Wbase + (size_t)n * 1024 + k0 + koff;
            u16* dst = Ws + r * 32 + koff;
            async_copy16(dst, src);
            async_copy16(dst + 8, src + 8);
        }
        // stage A (convert f32 -> bf16): 32x32, 4 elems/thread
        {
            int e0  = tid * 4;
            int row = e0 >> 5;
            int kk  = e0 & 31;
            int kg  = k0 + kk;
            const float* src = (kg < 512)
                ? (inp + (size_t)row * inp_stride + kg)
                : (hL  + (size_t)row * H_ + (kg - 512));
            u16* dst = As + row * 32 + kk;
            #pragma unroll
            for (int j = 0; j < 4; ++j) dst[j] = f2bf(src[j]);
        }
        async_wait0();
        __syncthreads();

        const int kof = (lane >= 16) ? 8 : 0;
        v16bf a = load_frag_lds(As + (mTile * 16 + (lane & 15)) * 32, kof);
        #pragma unroll
        for (int q = 0; q < 2; ++q) {
            int nT    = nPair * 2 + q;
            int strip = nT >> 1;
            int col   = (nT & 1) * 16 + (lane & 15);
            v16bf b = load_frag_lds(Ws + (strip * 32 + col) * 32, kof);
            acc[q] = __builtin_amdgcn_wmma_f32_16x16x32_bf16(
                false, a, false, b, (short)0, acc[q], false, false);
        }
        __syncthreads();
    }

    // spill gate fragments to LDS so each thread can see its i/f/g/o quad
    #pragma unroll
    for (int q = 0; q < 2; ++q) {
        int nT     = nPair * 2 + q;
        int colIdx = nT * 16 + (lane & 15);          // == strip*32 + colInStrip
        int rbase  = mTile * 16 + ((lane >= 16) ? 8 : 0);
        #pragma unroll
        for (int i = 0; i < 8; ++i)
            Gs[(rbase + i) * 128 + colIdx] = acc[q][i];
    }
    __syncthreads();

    // cell update + residual, 1024 (row,unit) pairs, 4 per thread
    #pragma unroll
    for (int it = 0; it < 4; ++it) {
        int idx = tid + 256 * it;
        int row = idx >> 5;                          // batch row
        int u   = idx & 31;                          // unit within block
        float ig = Gs[row * 128 +       u];
        float fg = Gs[row * 128 +  32 + u];
        float gg = Gs[row * 128 +  64 + u];
        float og = Gs[row * 128 +  96 + u];
        int j = j0 + u;
        size_t hidx = (size_t)row * H_ + j;
        float cold = cL[hidx];
        float cn = sigmoidf_(fg) * cold + sigmoidf_(ig) * tanhf(gg);
        float hn = sigmoidf_(og) * tanhf(cn);
        cL[hidx]  = cn;
        hLw[hidx] = hn;
        outp[(size_t)row * out_stride + j] = hn + inp[(size_t)row * inp_stride + j];
    }
}

// --------------------------------------------------------- classifier GEMM
// logits[2048, 32000] = outs[2048,512] @ W_cls^T ; 128x128 tile per block.
// Double-buffered LDS: B panel streams via async global->LDS while WMMAs run.
__global__ void cls_gemm_kernel(const float* __restrict__ outs,
                                const u16* __restrict__ Wb,   // [V][512] bf16
                                float* __restrict__ logits) {
    __shared__ u16 As[2][128 * 32];   // 16 KB
    __shared__ u16 Bs[2][128 * 32];   // 16 KB
    const int tid  = threadIdx.x;
    const int lane = tid & 31;
    const int wave = tid >> 5;
    const int mBase = blockIdx.x * 128;
    const int nBase = blockIdx.y * 128;
    const int r    = tid >> 1;            // 0..127
    const int koff = (tid & 1) * 16;

    v8f acc[8];
    #pragma unroll
    for (int q = 0; q < 8; ++q)
        #pragma unroll
        for (int i = 0; i < 8; ++i) acc[q][i] = 0.0f;

    const float* srcA0 = outs + (size_t)(mBase + r) * 512 + koff;
    const u16*   srcB0 = Wb   + (size_t)(nBase + r) * 512 + koff;

    // prologue: stage panel kb=0 into buffer 0
    {
        u16* dstA = As[0] + r * 32 + koff;
        #pragma unroll
        for (int j = 0; j < 16; ++j) dstA[j] = f2bf(srcA0[j]);
        u16* dstB = Bs[0] + r * 32 + koff;
        async_copy16(dstB,     srcB0);
        async_copy16(dstB + 8, srcB0 + 8);
    }
    async_wait0();
    __syncthreads();

    for (int kb = 0; kb < 16; ++kb) {
        const int cur = kb & 1;
        const int nxt = cur ^ 1;
        if (kb + 1 < 16) {
            const int k0n = (kb + 1) * 32;
            // A panel: f32 -> bf16 conversion through VALU
            const float* srcA = srcA0 + k0n;
            u16* dstA = As[nxt] + r * 32 + koff;
            #pragma unroll
            for (int j = 0; j < 16; ++j) dstA[j] = f2bf(srcA[j]);
            // B panel: async global -> LDS, overlaps with WMMAs below
            const u16* srcB = srcB0 + k0n;
            u16* dstB = Bs[nxt] + r * 32 + koff;
            async_copy16(dstB,     srcB);
            async_copy16(dstB + 8, srcB + 8);
        }

        const int kof = (lane >= 16) ? 8 : 0;
        v16bf a = load_frag_lds(As[cur] + (wave * 16 + (lane & 15)) * 32, kof);
        #pragma unroll
        for (int nT = 0; nT < 8; ++nT) {
            v16bf b = load_frag_lds(Bs[cur] + (nT * 16 + (lane & 15)) * 32, kof);
            acc[nT] = __builtin_amdgcn_wmma_f32_16x16x32_bf16(
                false, a, false, b, (short)0, acc[nT], false, false);
        }
        async_wait0();
        __syncthreads();
    }

    const int rbase = mBase + wave * 16 + ((lane >= 16) ? 8 : 0);
    #pragma unroll
    for (int nT = 0; nT < 8; ++nT) {
        int col = nBase + nT * 16 + (lane & 15);
        #pragma unroll
        for (int i = 0; i < 8; ++i)
            logits[(size_t)(rbase + i) * V_ + col] = acc[nT][i];
    }
}

// -------------------------------------------- log_softmax over time axis (T)
__global__ void logsoftmax_kernel(float* __restrict__ out,
                                  const float* __restrict__ b_cls) {
    int g = blockIdx.x * blockDim.x + threadIdx.x;   // 0 .. 32*32000
    int b = g / V_;
    int v = g - b * V_;
    float bias = b_cls[v];
    float vals[T_];
    float mx = -INFINITY;
    #pragma unroll
    for (int t = 0; t < T_; ++t) {
        float xv = out[((size_t)(b * T_ + t)) * V_ + v] + bias;
        vals[t] = xv;
        mx = fmaxf(mx, xv);
    }
    float s = 0.0f;
    #pragma unroll
    for (int t = 0; t < T_; ++t) s += __expf(vals[t] - mx);
    float lse = mx + __logf(s);
    #pragma unroll
    for (int t = 0; t < T_; ++t)
        out[((size_t)(b * T_ + t)) * V_ + v] = vals[t] - lse;
}

// ---------------------------------------------------------------- launcher
extern "C" void kernel_launch(void* const* d_in, const int* in_sizes, int n_in,
                              void* d_out, int out_size, void* d_ws, size_t ws_size,
                              hipStream_t stream) {
    (void)in_sizes; (void)n_in; (void)out_size; (void)ws_size;
    const int*   dec   = (const int*)d_in[1];
    const float* h0    = (const float*)d_in[2];
    const float* c0    = (const float*)d_in[3];
    const float* emb   = (const float*)d_in[4];
    const float* W_ih  = (const float*)d_in[5];
    const float* W_hh  = (const float*)d_in[6];
    const float* b_ih  = (const float*)d_in[7];
    const float* b_hh  = (const float*)d_in[8];
    const float* W_cls = (const float*)d_in[9];
    const float* b_cls = (const float*)d_in[10];

    char* ws = (char*)d_ws;
    u16*   Wl       = (u16*)(ws);                    // 16,777,216 B
    u16*   Wcb      = (u16*)(ws + 16777216);         // 32,768,000 B
    float* x        = (float*)(ws + 49545216);       //  4,194,304 B
    float* outs     = (float*)(ws + 53739520);       //  4,194,304 B
    float* hA       = (float*)(ws + 57933824);       //    262,144 B
    float* hB       = (float*)(ws + 58195968);       //    262,144 B
    float* cbuf     = (float*)(ws + 58458112);       //    262,144 B
    float* layerout = (float*)(ws + 58720256);       //    262,144 B  (total ~56.2 MiB)

    const size_t state_bytes = (size_t)L_ * B_ * H_ * sizeof(float);
    (void)hipMemcpyAsync(hA,   h0, state_bytes, hipMemcpyDeviceToDevice, stream);
    (void)hipMemcpyAsync(cbuf, c0, state_bytes, hipMemcpyDeviceToDevice, stream);

    embed_kernel<<<B_ * T_, 128, 0, stream>>>(emb, dec, x);
    prep_wlstm<<<(L_ * 2048 * 1024 / 4) / 256, 256, 0, stream>>>(W_ih, W_hh, Wl);
    prep_wcls<<<((size_t)V_ * 512 / 4) / 256, 256, 0, stream>>>(W_cls, Wcb);

    for (int t = 0; t < T_; ++t) {
        float* hin  = (t & 1) ? hB : hA;
        float* hout = (t & 1) ? hA : hB;
        for (int l = 0; l < L_; ++l) {
            const float* inp; int istr; float* op; int ostr;
            if (l == 0) { inp = x + (size_t)t * H_;                   istr = T_ * H_; }
            else        { inp = layerout + (size_t)(l - 1) * B_ * H_; istr = H_;      }
            if (l == 3) { op = outs + (size_t)t * H_;                 ostr = T_ * H_; }
            else        { op = layerout + (size_t)l * B_ * H_;        ostr = H_;      }
            lstm_layer_kernel<<<16, 256, 0, stream>>>(Wl, b_ih, b_hh, hin, hout, cbuf,
                                                      inp, istr, op, ostr, l);
        }
    }

    float* outPtr = (float*)d_out;
    // tuple layout: out [B,T,V] | hT [L,B,H] | cT [L,B,H]
    // final h written at t=63 (odd) into hA
    (void)hipMemcpyAsync(outPtr + (size_t)B_ * T_ * V_,                hA,   state_bytes,
                         hipMemcpyDeviceToDevice, stream);
    (void)hipMemcpyAsync(outPtr + (size_t)B_ * T_ * V_ + L_ * B_ * H_, cbuf, state_bytes,
                         hipMemcpyDeviceToDevice, stream);

    cls_gemm_kernel<<<dim3(B_ * T_ / 128, V_ / 128), 256, 0, stream>>>(outs, Wcb, outPtr);
    logsoftmax_kernel<<<(B_ * V_) / 256, 256, 0, stream>>>(outPtr, b_cls);
}
